// GNN_Layer_31353261260808
// MI455X (gfx1250) — compile-verified, matrix operand order checked by reading, not run on hardware
//
#include <hip/hip_runtime.h>
#include <hip/hip_bf16.h>
#include <math.h>

// GATConv forward for MI455X (gfx1250, wave32).
// Pipeline (all on `stream`):
//   0) init:   out = 0, denom = 0
//   1) gemm:   xph = f16(x @ W)      (LDS-staged f16 tiles -> v_wmma_f32_16x16x32_f16)
//   2) logits: a_s/a_d = <xph, att>  per (node, head), f32
//   3) eagg:   ONE edge pass: e2 = exp(leakyrelu(a_s[src]+a_d[dst]))  (max-shift dropped:
//              mathematically identical alpha, logits are O(1) here)
//              denom[dst,h] += e2[h];  out[dst,:] += e2[h]*xph[src,:]  (global_atomic_add_f32)
//   4) final:  out = relu(out/denom + bias)  in place

#define HEADS 4
#define FPH   32
#define CIN   128
#define COUT  128   // HEADS*FPH
#define NEG_SLOPE 0.2f
#define LPITCH 130  // LDS pitch in halves (spreads banks; keeps 4B alignment for even cols)

typedef __attribute__((ext_vector_type(16))) _Float16 v16h;
typedef __attribute__((ext_vector_type(8)))  float    v8f;

// ---------------------------------------------------------------- utilities
__device__ __forceinline__ float leaky(float v) {
    return v > 0.0f ? v : NEG_SLOPE * v;
}

// ---------------------------------------------------------------- kernel 0
__global__ void gat_init(float* __restrict__ out, float* __restrict__ denom, int n) {
    int i = blockIdx.x * blockDim.x + threadIdx.x;
    if (i < n * COUT) out[i] = 0.0f;
    if (i < n * HEADS) denom[i] = 0.0f;
}

// ---------------------------------------------------------------- kernel 1
// xph = f16(x @ W).  Block = 256 threads = 8 waves.  Block b covers rows
// [16b,16b+16); wave w covers output columns [16w,16w+16).  The block stages
// the 16x128 A tile and the full (transposed) 128x128 W in LDS as f16 once,
// then each wave builds WMMA fragments with aligned 4B DS reads.
//
// Fragment layouts per CDNA5 ISA 7.12.2 (wave32):
//   A 16x32 f16 : lane L holds row (L&15); element idx 2v+j holds
//                 K = (v<4?0:16) + (L>=16?8:0) + 2*(v&3) + j
//   B 32x16 f16 : symmetric to A with lane striding over N
//   C/D 16x16 f32: reg r -> M = r + (L>=16?8:0), N = (L&15)
__global__ __launch_bounds__(256) void gat_gemm(const float* __restrict__ x,
                                                const float* __restrict__ W,
                                                _Float16* __restrict__ xph, int n) {
    __shared__ _Float16 sW[COUT * LPITCH];  // sW[col*LPITCH + k] = f16(W[k][col])
    __shared__ _Float16 sA[16 * LPITCH];    // sA[r*LPITCH + c]  = f16(x[row0+r][c])

    const int tid  = threadIdx.x;
    const int lane = tid & 31;
    const int wave = tid >> 5;
    const int row0 = blockIdx.x * 16;
    const int col0 = wave * 16;
    const int l15  = lane & 15;
    const int kh   = (lane >> 4) * 8;   // K sub-offset 0 or 8

    // Cooperative stage: W transposed into LDS (f16)
    for (int j = tid; j < CIN * COUT; j += 256) {
        const int k = j >> 7, c = j & (COUT - 1);
        sW[c * LPITCH + k] = (_Float16)W[j];
    }
    // Cooperative stage: A tile into LDS (f16), row-clamped (n%16==0 normally)
    for (int j = tid; j < 16 * CIN; j += 256) {
        const int r = j >> 7, c = j & (CIN - 1);
        int gr = row0 + r;
        if (gr >= n) gr = n - 1;
        sA[r * LPITCH + c] = (_Float16)x[(size_t)gr * CIN + c];
    }
    __syncthreads();

    v8f acc = {};
#pragma unroll
    for (int kc = 0; kc < CIN; kc += 32) {
        v16h a, b;
        unsigned int* au = (unsigned int*)&a;
        unsigned int* bu = (unsigned int*)&b;
        const unsigned int* pa =
            (const unsigned int*)(sA + l15 * LPITCH + kc + kh);
        const unsigned int* pb =
            (const unsigned int*)(sW + (size_t)(col0 + l15) * LPITCH + kc + kh);
#pragma unroll
        for (int v = 0; v < 8; ++v) {
            const int kk = ((v < 4) ? 0 : 16) + 2 * (v & 3);  // even -> 4B aligned
            au[v] = pa[kk >> 1];
            bu[v] = pb[kk >> 1];
        }
        acc = __builtin_amdgcn_wmma_f32_16x16x32_f16(
            /*neg_a=*/false, a, /*neg_b=*/false, b,
            /*c_mod=*/(short)0, acc, /*reuse_a=*/false, /*reuse_b=*/false);
    }

    // Epilogue: uniform fast path (always taken when n%16==0) -> straight-line
    // unconditional b16 stores; divergent per-row guards only on a ragged tail.
    const int moff = (lane >> 4) * 8;
    _Float16* op = xph + (size_t)(row0 + moff) * COUT + col0 + l15;
    if (row0 + 16 <= n) {
#pragma unroll
        for (int r = 0; r < 8; ++r) {
            op[(size_t)r * COUT] = (_Float16)acc[r];
        }
    } else {
#pragma unroll
        for (int r = 0; r < 8; ++r) {
            if (row0 + moff + r < n) op[(size_t)r * COUT] = (_Float16)acc[r];
        }
    }
}

// ---------------------------------------------------------------- kernel 2
// a_s[n,h] = sum_f xph[n,h,f]*att_src[h,f];  a_d likewise (f32 accumulate).
__global__ void gat_logits(const _Float16* __restrict__ xph,
                           const float* __restrict__ att_src,
                           const float* __restrict__ att_dst,
                           float* __restrict__ a_s, float* __restrict__ a_d, int n) {
    int i = blockIdx.x * blockDim.x + threadIdx.x;
    if (i >= n * HEADS) return;
    const int h    = i & (HEADS - 1);
    const int node = i >> 2;
    const _Float16* v = xph + (size_t)node * COUT + h * FPH;
    float s = 0.0f, d = 0.0f;
#pragma unroll
    for (int f = 0; f < FPH; ++f) {
        const float xv = (float)v[f];
        s += xv * att_src[h * FPH + f];
        d += xv * att_dst[h * FPH + f];
    }
    a_s[i] = s;
    a_d[i] = d;
}

// ---------------------------------------------------------------- kernel 3
// Single fused edge pass, one wave32 per edge (incl. self loops):
//   e2[h]        = exp(leakyrelu(a_s[src,h] + a_d[dst,h]))
//   denom[dst,h] += e2[h]                 (lanes 0..3)
//   out[dst,h*32+f] += e2[h]*xph[src,h*32+f]   (lane f, all 4 heads)
// Normalization by denom is deferred to the finalize kernel.
__global__ __launch_bounds__(256) void gat_edge_agg(
        const int* __restrict__ srcI, const int* __restrict__ dstI, int E, int n,
        const float* __restrict__ a_s, const float* __restrict__ a_d,
        float* __restrict__ denom,
        const _Float16* __restrict__ xph, float* __restrict__ out) {
    const int gid = blockIdx.x * blockDim.x + threadIdx.x;
    const int e   = gid >> 5;
    const int f   = gid & 31;
    if (e >= E + n) return;
    const int s = (e < E) ? srcI[e] : (e - E);
    const int d = (e < E) ? dstI[e] : (e - E);
    // Per-edge scalars: same address across the wave -> broadcast loads.
    const float4 as = *(const float4*)(a_s + (size_t)s * HEADS);
    const float4 ad = *(const float4*)(a_d + (size_t)d * HEADS);
    float e2[HEADS];
    e2[0] = expf(leaky(as.x + ad.x));
    e2[1] = expf(leaky(as.y + ad.y));
    e2[2] = expf(leaky(as.z + ad.z));
    e2[3] = expf(leaky(as.w + ad.w));
    if (f < HEADS) atomicAdd(denom + (size_t)d * HEADS + f, e2[f]);
    const _Float16* xs = xph + (size_t)s * COUT;
    float* od = out + (size_t)d * COUT;
#pragma unroll
    for (int h = 0; h < HEADS; ++h) {
        atomicAdd(od + h * FPH + f, e2[h] * (float)xs[h * FPH + f]);
    }
}

// ---------------------------------------------------------------- kernel 4
// out = relu(out/denom + bias)
__global__ void gat_finalize(float* __restrict__ out, const float* __restrict__ denom,
                             const float* __restrict__ bias, int total) {
    int i = blockIdx.x * blockDim.x + threadIdx.x;
    if (i >= total) return;
    const int node = i >> 7;              // /COUT
    const int h    = (i >> 5) & (HEADS - 1);
    const float v  = out[i] / denom[node * HEADS + h] + bias[i & (COUT - 1)];
    out[i] = v > 0.0f ? v : 0.0f;
}

// ---------------------------------------------------------------- launcher
extern "C" void kernel_launch(void* const* d_in, const int* in_sizes, int n_in,
                              void* d_out, int out_size, void* d_ws, size_t ws_size,
                              hipStream_t stream) {
    const float* x       = (const float*)d_in[0];
    const int*   ei      = (const int*)d_in[1];
    const float* W       = (const float*)d_in[2];
    const float* att_src = (const float*)d_in[3];
    const float* att_dst = (const float*)d_in[4];
    const float* bias    = (const float*)d_in[5];
    float*       out     = (float*)d_out;

    const int n  = in_sizes[0] / CIN;  // 50000
    const int E  = in_sizes[1] / 2;    // 800000
    const int ET = E + n;              // edges incl. self loops
    const int* srcI = ei;
    const int* dstI = ei + E;

    // Workspace layout: xph f16[n*128] | a_s f32[n*4] | a_d f32[n*4] | denom f32[n*4]
    _Float16* xph  = (_Float16*)d_ws;
    float*    fws  = (float*)(xph + (size_t)n * COUT);
    float*    a_s  = fws;
    float*    a_d  = a_s + (size_t)n * HEADS;
    float*    denom = a_d + (size_t)n * HEADS;

    const int total = n * COUT;

    gat_init<<<(total + 255) / 256, 256, 0, stream>>>(out, denom, n);
    gat_gemm<<<(n + 15) / 16, 256, 0, stream>>>(x, W, xph, n);
    gat_logits<<<(n * HEADS + 255) / 256, 256, 0, stream>>>(xph, att_src, att_dst, a_s, a_d, n);
    const long long aggThreads = (long long)ET * 32;
    gat_edge_agg<<<(unsigned int)((aggThreads + 255) / 256), 256, 0, stream>>>(
        srcI, dstI, E, n, a_s, a_d, denom, xph, out);
    gat_finalize<<<(total + 255) / 256, 256, 0, stream>>>(out, denom, bias, total);
}
